// BSRNN_69887707841230
// MI455X (gfx1250) — matrix operand order repaired and checked by reading, not compiled, not run on hardware
//
#include <hip/hip_runtime.h>

typedef __bf16 bf16;
typedef __attribute__((ext_vector_type(16))) __bf16 v16bf;
typedef __attribute__((ext_vector_type(8)))  __bf16 v8bf;
typedef __attribute__((ext_vector_type(8)))  float  v8f;

#define NBANDS 29
#define NB 64
#define TT 1000
#define BB 2
#define KB 29
#define MTOT (BB*TT*KB)   /* 58000 rows of (b,t,k) */
#define FREQ 2049

struct BandPtrs {
  const float* w1[NBANDS]; const float* b1[NBANDS];
  const float* w2[NBANDS]; const float* b2[NBANDS];
  int bsz[NBANDS]; int boff[NBANDS];
};

struct ScanArgs {
  const bf16* x;
  long long x_outer, x_inner, x_step;   // row base = (r/rpo)*outer + (r%rpo)*inner ; step stride in elems
  int rpo;
  const float *wih, *whh, *bih, *bhh;   // f32 weights (out,in) row-major
  bf16* out; int outD; int outOff;      // out[(r*S+s)*outD + outOff + col]
  int R, S, rev;
};

// ---------------- WMMA helpers (CDNA5 bf16 16x16x32, f32 accum) ----------------

__device__ __forceinline__ v8f wmma_bf(v16bf a, v16bf b, v8f c) {
  return __builtin_amdgcn_wmma_f32_16x16x32_bf16(false, a, false, b, (short)0, c, false, false);
}

__device__ __forceinline__ v8f splat8(float v) {
  v8f r;
#pragma unroll
  for (int i = 0; i < 8; ++i) r[i] = v;
  return r;
}

__device__ __forceinline__ float sigf(float x) { return 1.0f / (1.0f + __expf(-x)); }
__device__ __forceinline__ float tanh_(float x) { return 2.0f * sigf(2.0f * x) - 1.0f; }

union V16U { v16bf v; v8bf h[2]; };

// A fragment (16x32, row = lane&15 supplied via per-lane row pointer).
// Layout: lanes 0-15 hold K 0-15, lanes 16-31 hold K 16-31 (matches B fragments).
__device__ __forceinline__ v16bf load_a(const bf16* rowp, int kbase) {
  const int lane = threadIdx.x & 31;
  const bf16* p = rowp + kbase + ((lane >> 4) << 4);
  V16U u;
  u.h[0] = *(const v8bf*)p;
  u.h[1] = *(const v8bf*)(p + 8);
  return u.v;
}

// B fragment from bf16 (LDS-staged) weights W[n][k] row-major, full tiles only.
__device__ __forceinline__ v16bf load_b_bf16(const bf16* w, int ld, int nbase, int kbase) {
  const int lane = threadIdx.x & 31;
  const bf16* p = w + (nbase + (lane & 15)) * ld + kbase + ((lane >> 4) << 4);
  V16U u;
  u.h[0] = *(const v8bf*)p;
  u.h[1] = *(const v8bf*)(p + 8);
  return u.v;
}

// Unguarded B fragment straight from f32 weights (full tiles; used only outside hot loops).
__device__ __forceinline__ v16bf load_b_f32x(const float* w, int ld, int nbase, int kbase) {
  const int lane = threadIdx.x & 31;
  const float* p = w + (long long)(nbase + (lane & 15)) * ld + kbase + ((lane >> 4) << 4);
  v16bf b;
#pragma unroll
  for (int i = 0; i < 16; ++i) b[i] = (bf16)p[i];
  return b;
}

// ---------------- Band-split FC: h[b,t,band,:] = leaky(x_band @ w1.T + b1) @ w2.T + b2 ----

__global__ __launch_bounds__(64) void band_fc_kernel(const float* __restrict__ xr,
                                                     const float* __restrict__ xi,
                                                     BandPtrs P, bf16* __restrict__ h_out) {
  const int band = blockIdx.y, bi = blockIdx.z;
  const int b = P.bsz[band], off = P.boff[band];
  const int K2 = 2 * b;
  const int t0 = blockIdx.x * 32;
  const int tid = threadIdx.x, wave = tid >> 5, lane = tid & 31, m = lane & 15;
  const int rofs = (lane >> 4) << 3;
  __shared__ bf16 aT[32 * 32];
  __shared__ bf16 w1s[64 * 32];
  __shared__ bf16 w2s[64 * 64];
  __shared__ bf16 y1[32 * 64];

  // stage w2 (full 64x64) once; first use is after >=1 barrier below
  {
    const float* w2 = P.w2[band];
    for (int e = tid; e < 64 * 64; e += 64) w2s[e] = (bf16)w2[e];
  }

  v8f c1[4];
#pragma unroll
  for (int nt = 0; nt < 4; ++nt) c1[nt] = splat8(0.0f);

  for (int k0 = 0; k0 < K2; k0 += 32) {
    // Stage a 32x32 chunk of the (transposed, concat[re|im]) band input (branchless selects).
    for (int e = tid; e < 32 * 32; e += 64) {
      int mm = e >> 5, cc = (e & 31) + k0;
      int t = t0 + mm; t = (t < TT) ? t : (TT - 1);
      int ccl = (cc < K2) ? cc : (K2 - 1);
      int inb = (ccl < b);
      int f = inb ? (off + ccl) : (off + ccl - b);
      const float* src = inb ? xr : xi;
      float v = src[((long long)(bi * FREQ + f)) * TT + t];
      v = (cc < K2) ? v : 0.0f;
      aT[mm * 32 + (e & 31)] = (bf16)v;
    }
    // Stage this K-chunk of w1 (64 x 32), zero-padded past K2 (branchless).
    {
      const float* w1 = P.w1[band];
      for (int e = tid; e < 64 * 32; e += 64) {
        int n = e >> 5, kk = (e & 31) + k0;
        int kkl = (kk < K2) ? kk : (K2 - 1);
        float v = w1[(long long)n * K2 + kkl];
        w1s[e] = (kk < K2) ? (bf16)v : (bf16)0.0f;
      }
    }
    __syncthreads();
    v16bf a = load_a(aT + (wave * 16 + m) * 32, 0);
#pragma unroll
    for (int nt = 0; nt < 4; ++nt)
      c1[nt] = wmma_bf(a, load_b_bf16(w1s, 32, nt * 16, 0), c1[nt]);
    __syncthreads();
  }

  // bias + leaky, stage y1 (wave-local 16x64 rows) as bf16
#pragma unroll
  for (int nt = 0; nt < 4; ++nt) {
    const int col = nt * 16 + m;
    const float bias = P.b1[band][col];
#pragma unroll
    for (int r = 0; r < 8; ++r) {
      float v = c1[nt][r] + bias;
      v = (v > 0.0f) ? v : 0.01f * v;
      y1[(wave * 16 + rofs + r) * 64 + col] = (bf16)v;
    }
  }
  // second GEMM (K=64)
  v8f c2[4];
#pragma unroll
  for (int nt = 0; nt < 4; ++nt) c2[nt] = splat8(0.0f);
#pragma unroll
  for (int kt = 0; kt < 2; ++kt) {
    v16bf a2 = load_a(y1 + (wave * 16 + m) * 64, kt * 32);
#pragma unroll
    for (int nt = 0; nt < 4; ++nt)
      c2[nt] = wmma_bf(a2, load_b_bf16(w2s, NB, nt * 16, kt * 32), c2[nt]);
  }
#pragma unroll
  for (int nt = 0; nt < 4; ++nt) {
    const int col = nt * 16 + m;
    const float bias = P.b2[band][col];
#pragma unroll
    for (int r = 0; r < 8; ++r) {
      int t = t0 + wave * 16 + rofs + r;
      if (t < TT)
        h_out[((long long)(bi * TT + t) * KB + band) * NB + col] = (bf16)(c2[nt][r] + bias);
    }
  }
}

// ---------------- Generic position-wise FC (N=64): fc_in / fc_out (+residual) -----------
// 128 threads = 4 waves x 16 rows -> 64 rows/block; LDS weight staging amortized 2x.

__global__ __launch_bounds__(128) void fc_kernel(const bf16* __restrict__ x,
                                                 const float* __restrict__ w,
                                                 const float* __restrict__ bias,
                                                 const bf16* __restrict__ res,
                                                 bf16* __restrict__ out,
                                                 int M, int Kd, int map_mode) {
  __shared__ bf16 w_s[64 * 128];
  const int tid = threadIdx.x, wave = tid >> 5, lane = tid & 31, m = lane & 15;
  // stage weights (64 x Kd) as bf16 once per block
  for (int e = tid; e < 64 * Kd; e += 128) w_s[e] = (bf16)w[e];
  __syncthreads();

  const int pt = blockIdx.x * 64 + wave * 16;
  int p = pt + m; p = (p < M) ? p : (M - 1);
  long long in_base;
  if (map_mode == 0) {
    in_base = (long long)p * Kd;
  } else {  // odd-module fc_out: output pos (b,t,k) <- lstm row (b,k), step t
    int bq = p / (TT * KB);
    int t = (p / KB) % TT;
    int k = p % KB;
    in_base = ((long long)(bq * KB + k) * TT + t) * Kd;
  }
  const bf16* rowp = x + in_base;
  v8f acc[4];
#pragma unroll
  for (int nt = 0; nt < 4; ++nt) acc[nt] = splat8(0.0f);
  for (int kt = 0; kt < (Kd >> 5); ++kt) {
    v16bf a = load_a(rowp, kt * 32);
#pragma unroll
    for (int nt = 0; nt < 4; ++nt)
      acc[nt] = wmma_bf(a, load_b_bf16(w_s, Kd, nt * 16, kt * 32), acc[nt]);
  }
  const int rofs = (lane >> 4) << 3;
#pragma unroll
  for (int nt = 0; nt < 4; ++nt) {
    const int col = nt * 16 + m;
    const float bv = bias[col];
#pragma unroll
    for (int r = 0; r < 8; ++r) {
      int rg = pt + rofs + r;
      if (rg < M) {
        float v = acc[nt][r] + bv;
        if (res) v += (float)res[(long long)rg * NB + col];
        out[(long long)rg * NB + col] = (bf16)v;
      }
    }
  }
}

// ---------------- Persistent LSTM scan: wave owns 16 rows, c-state in WMMA frags --------
// All weights staged to LDS bf16 once; the step loop feeds WMMA purely from LDS/registers,
// and prefetches the next timestep's activations (global_prefetch) to hide L2 latency.

template <int IN_DIM>
__global__ __launch_bounds__(128) void lstm_scan(ScanArgs A) {
  constexpr int NKX = IN_DIM / 32;
  __shared__ bf16 wih_s[256 * IN_DIM];  // 32/64 KB
  __shared__ bf16 whh_s[256 * 64];      // 32 KB
  __shared__ bf16 hscr[4 * 16 * 64];    // 8 KB per-wave h recycle tiles
  const int tid = threadIdx.x, wave = tid >> 5, lane = tid & 31, m = lane & 15;
  const int rofs = (lane >> 4) << 3;

  for (int e = tid; e < 256 * IN_DIM; e += 128) wih_s[e] = (bf16)A.wih[e];
  for (int e = tid; e < 256 * 64; e += 128) whh_s[e] = (bf16)A.whh[e];
  bf16* hw = hscr + wave * (16 * 64);
  for (int e = lane; e < 16 * 64; e += 32) hw[e] = (bf16)0.0f;
  __syncthreads();

  const int row0 = blockIdx.x * 64 + wave * 16;
  int r = row0 + m; r = (r < A.R) ? r : (A.R - 1);
  const bf16* xrow = A.x + (long long)(r / A.rpo) * A.x_outer + (long long)(r % A.rpo) * A.x_inner;
  const bf16* hrowp = hw + m * 64;

  float biasv[16];
#pragma unroll
  for (int nt = 0; nt < 16; ++nt) biasv[nt] = A.bih[nt * 16 + m] + A.bhh[nt * 16 + m];

  v8f cst[4];
#pragma unroll
  for (int q = 0; q < 4; ++q) cst[q] = splat8(0.0f);

  for (int s = 0; s < A.S; ++s) {
    const int ss = A.rev ? (A.S - 1 - s) : s;
    const bf16* xs = xrow + (long long)ss * A.x_step;
    // prefetch next step's row while this step computes
    if (s + 1 < A.S) {
      const int sn = A.rev ? (A.S - 2 - s) : (s + 1);
      __builtin_prefetch((const void*)(xrow + (long long)sn * A.x_step), 0, 3);
    }
    v16bf xa[NKX];
#pragma unroll
    for (int k = 0; k < NKX; ++k) xa[k] = load_a(xs, k * 32);
    v16bf ha[2];
#pragma unroll
    for (int k = 0; k < 2; ++k) ha[k] = load_a(hrowp, k * 32);

    v8f g[16];
#pragma unroll
    for (int nt = 0; nt < 16; ++nt) {
      g[nt] = splat8(biasv[nt]);
#pragma unroll
      for (int k = 0; k < NKX; ++k)
        g[nt] = wmma_bf(xa[k], load_b_bf16(wih_s, IN_DIM, nt * 16, k * 32), g[nt]);
#pragma unroll
      for (int k = 0; k < 2; ++k)
        g[nt] = wmma_bf(ha[k], load_b_bf16(whh_s, 64, nt * 16, k * 32), g[nt]);
    }
    // gates: cols [0:64)=i [64:128)=f [128:192)=g [192:256)=o
#pragma unroll
    for (int q = 0; q < 4; ++q) {
      const int col = q * 16 + m;
#pragma unroll
      for (int e = 0; e < 8; ++e) {
        float iv = sigf(g[q][e]);
        float fv = sigf(g[4 + q][e]);
        float gv = tanh_(g[8 + q][e]);
        float ov = sigf(g[12 + q][e]);
        float cv = fv * cst[q][e] + iv * gv;
        cst[q][e] = cv;
        float hv = ov * tanh_(cv);
        const int row = rofs + e;
        hw[row * 64 + col] = (bf16)hv;            // same-wave DS is in-order: safe recycle
        const int rg = row0 + row;
        if (rg < A.R)
          A.out[((long long)rg * A.S + ss) * A.outD + A.outOff + col] = (bf16)hv;
      }
    }
  }
}

// ---------------- Band-back FC + GLU -> masks stored transposed as (B,F,T) --------------

__global__ __launch_bounds__(64) void band_back_kernel(const bf16* __restrict__ h, BandPtrs P,
                                                       float* __restrict__ mr,
                                                       float* __restrict__ mi) {
  const int band = blockIdx.y, bi = blockIdx.z;
  const int b = P.bsz[band], off = P.boff[band];
  const int twob = 2 * b, b4 = 4 * b;
  const int t0 = blockIdx.x * 32;
  const int tid = threadIdx.x, wave = tid >> 5, lane = tid & 31, m = lane & 15;
  const int rofs = (lane >> 4) << 3;
  __shared__ bf16 y1[32 * 256];        // 16 KB
  __shared__ bf16 wa_s[16 * 256];      // 8 KB: w2 rows for "value" half
  __shared__ bf16 wb_s[16 * 256];      // 8 KB: w2 rows for "gate" half
  __shared__ float tbuf[2][16 * 16];   // 2 KB

  int tr = t0 + wave * 16 + m; tr = (tr < TT) ? tr : (TT - 1);
  const bf16* hrow = h + ((long long)(bi * TT + tr) * KB + band) * NB;

  // y1 = leaky(h @ w1.T + b1), N=256 (w1 is full 256x64: unguarded f32 fragments, used once)
#pragma unroll
  for (int nt = 0; nt < 16; ++nt) {
    const int col = nt * 16 + m;
    v8f a0 = splat8(P.b1[band][col]);
#pragma unroll
    for (int k = 0; k < 2; ++k)
      a0 = wmma_bf(load_a(hrow, k * 32), load_b_f32x(P.w1[band], NB, nt * 16, k * 32), a0);
#pragma unroll
    for (int r = 0; r < 8; ++r) {
      float v = a0[r];
      v = (v > 0.0f) ? v : 0.01f * v;
      y1[(wave * 16 + rofs + r) * 256 + col] = (bf16)v;
    }
  }

  const bf16* yrow = y1 + (wave * 16 + m) * 256;
  float* tb = tbuf[wave];
  const int nct = (twob + 15) >> 4;
  const float* w2 = P.w2[band];
  for (int ct = 0; ct < nct; ++ct) {
    const int cb = ct * 16;
    // cooperatively stage the two ragged 16x256 row-chunks of w2 as bf16 (branchless)
    __syncthreads();
    for (int e = tid; e < 16 * 256; e += 64) {
      int rr = e >> 8, k = e & 255;
      int ra = cb + rr, rb = twob + cb + rr;
      int ral = (ra < b4) ? ra : (b4 - 1);
      int rbl = (rb < b4) ? rb : (b4 - 1);
      float va = w2[(long long)ral * 256 + k];
      float vb = w2[(long long)rbl * 256 + k];
      wa_s[e] = (ra < b4) ? (bf16)va : (bf16)0.0f;
      wb_s[e] = (rb < b4) ? (bf16)vb : (bf16)0.0f;
    }
    __syncthreads();
    v8f ya = splat8(0.0f), yb = splat8(0.0f);
#pragma unroll
    for (int k = 0; k < 8; ++k) {
      v16bf a = load_a(yrow, k * 32);
      ya = wmma_bf(a, load_b_bf16(wa_s, 256, 0, k * 32), ya);
      yb = wmma_bf(a, load_b_bf16(wb_s, 256, 0, k * 32), yb);
    }
    int ca = cb + m, cg = twob + cb + m;
    float ba = P.b2[band][(ca < twob) ? ca : (twob - 1)];
    float bg = P.b2[band][(cg < b4) ? cg : (b4 - 1)];
    // GLU, write to wave-local transpose buffer [col][row]
#pragma unroll
    for (int r = 0; r < 8; ++r)
      tb[m * 16 + rofs + r] = (ya[r] + ba) * sigf(yb[r] + bg);
    // coalesced store: lane -> (col = lane>>1, 8 consecutive t's)
    const int c = cb + (lane >> 1);
    const int seg = lane & 1;
    if (c < twob) {
      const int f = off + ((c < b) ? c : (c - b));
      float* dst = (c < b) ? mr : mi;
      long long base = (long long)(bi * FREQ + f) * TT;
#pragma unroll
      for (int i = 0; i < 8; ++i) {
        int t = t0 + wave * 16 + seg * 8 + i;
        if (t < TT) dst[base + t] = tb[(lane >> 1) * 16 + seg * 8 + i];
      }
    }
  }
}

// ---------------- Final complex mix, fully coalesced over (B,F,T) -----------------------

__global__ void final_mix(const float* __restrict__ xr, const float* __restrict__ xi,
                          const float* __restrict__ mr, const float* __restrict__ mi,
                          float* __restrict__ out, int n) {
  int i = blockIdx.x * blockDim.x + threadIdx.x;
  if (i < n) {
    float vr = xr[i], vi = xi[i], a = mr[i], c = mi[i];
    out[2 * i] = vr * a - vi * c;
    out[2 * i + 1] = vr * c + vi * a;
  }
}

// ---------------- Host orchestration ----------------------------------------------------

extern "C" void kernel_launch(void* const* d_in, const int* in_sizes, int n_in,
                              void* d_out, int out_size, void* d_ws, size_t ws_size,
                              hipStream_t stream) {
  (void)in_sizes; (void)n_in; (void)out_size; (void)ws_size;
  static const int BANDS_H[NBANDS] = {5,5,5,5,10,10,10,10,10,10,10,10,50,50,50,50,50,50,50,50,
                                      100,100,100,100,100,250,250,250,299};
  int offs[NBANDS]; { int a = 0; for (int i = 0; i < NBANDS; ++i) { offs[i] = a; a += BANDS_H[i]; } }

  int idx = 0;
  const float* xr = (const float*)d_in[idx++];
  const float* xi = (const float*)d_in[idx++];

  BandPtrs bfc{};
  for (int i = 0; i < NBANDS; ++i) {
    bfc.w1[i] = (const float*)d_in[idx++]; bfc.b1[i] = (const float*)d_in[idx++];
    bfc.w2[i] = (const float*)d_in[idx++]; bfc.b2[i] = (const float*)d_in[idx++];
    bfc.bsz[i] = BANDS_H[i]; bfc.boff[i] = offs[i];
  }
  struct DirW { const float *wih, *whh, *bih, *bhh; };
  struct ModW { const float *fiw, *fib, *fw, *fb; DirW d[2][2]; int ndir; } mods[8];
  for (int j = 0; j < 8; ++j) {
    mods[j].ndir = (j % 2 == 0) ? 2 : 1;
    mods[j].fiw = (const float*)d_in[idx++]; mods[j].fib = (const float*)d_in[idx++];
    for (int l = 0; l < 2; ++l)
      for (int dd = 0; dd < mods[j].ndir; ++dd) {
        mods[j].d[l][dd].wih = (const float*)d_in[idx++];
        mods[j].d[l][dd].whh = (const float*)d_in[idx++];
        mods[j].d[l][dd].bih = (const float*)d_in[idx++];
        mods[j].d[l][dd].bhh = (const float*)d_in[idx++];
      }
    mods[j].fw = (const float*)d_in[idx++]; mods[j].fb = (const float*)d_in[idx++];
  }
  BandPtrs bbk{};
  for (int i = 0; i < NBANDS; ++i) {
    bbk.w1[i] = (const float*)d_in[idx++]; bbk.b1[i] = (const float*)d_in[idx++];
    bbk.w2[i] = (const float*)d_in[idx++]; bbk.b2[i] = (const float*)d_in[idx++];
    bbk.bsz[i] = BANDS_H[i]; bbk.boff[i] = offs[i];
  }

  // Workspace carve-up (all L2-resident: ~85 MB)
  char* ws = (char*)d_ws;
  size_t woff = 0;
  auto take = [&](size_t bytes) -> char* {
    char* p = ws + woff; woff = (woff + bytes + 255) & ~(size_t)255; return p;
  };
  bf16* h_a = (bf16*)take((size_t)MTOT * NB * 2);
  bf16* h_b = (bf16*)take((size_t)MTOT * NB * 2);
  bf16* z   = (bf16*)take((size_t)MTOT * NB * 2);
  bf16* l1  = (bf16*)take((size_t)MTOT * 128 * 2);
  bf16* l2  = (bf16*)take((size_t)MTOT * 128 * 2);
  float* mr = (float*)take((size_t)BB * FREQ * TT * 4);
  float* mi = (float*)take((size_t)BB * FREQ * TT * 4);

  dim3 bgrid(32, NBANDS, BB);
  band_fc_kernel<<<bgrid, 64, 0, stream>>>(xr, xi, bfc, h_a);

  const int fcgrid = (MTOT + 63) / 64;
  bf16 *hc = h_a, *hn = h_b;
  for (int j = 0; j < 8; ++j) {
    fc_kernel<<<fcgrid, 128, 0, stream>>>(hc, mods[j].fiw, mods[j].fib,
                                          (const bf16*)nullptr, z, MTOT, NB, 0);
    if (mods[j].ndir == 2) {  // even: bidir over K=29, 2000 independent rows
      ScanArgs a{};
      a.R = BB * TT; a.S = KB; a.rpo = BB * TT;
      a.x = z; a.x_outer = 0; a.x_inner = (long long)KB * NB; a.x_step = NB;
      a.out = l1; a.outD = 128;
      for (int dd = 0; dd < 2; ++dd) {
        a.wih = mods[j].d[0][dd].wih; a.whh = mods[j].d[0][dd].whh;
        a.bih = mods[j].d[0][dd].bih; a.bhh = mods[j].d[0][dd].bhh;
        a.outOff = dd * 64; a.rev = dd;
        lstm_scan<64><<<(a.R + 63) / 64, 128, 0, stream>>>(a);
      }
      a.x = l1; a.x_outer = 0; a.x_inner = (long long)KB * 128; a.x_step = 128;
      a.out = l2; a.outD = 128;
      for (int dd = 0; dd < 2; ++dd) {
        a.wih = mods[j].d[1][dd].wih; a.whh = mods[j].d[1][dd].whh;
        a.bih = mods[j].d[1][dd].bih; a.bhh = mods[j].d[1][dd].bhh;
        a.outOff = dd * 64; a.rev = dd;
        lstm_scan<128><<<(a.R + 63) / 64, 128, 0, stream>>>(a);
      }
      fc_kernel<<<fcgrid, 128, 0, stream>>>(l2, mods[j].fw, mods[j].fb, hc, hn, MTOT, 128, 0);
    } else {  // odd: unidir over T=1000, 58 rows (b,k); strided view of z, no transpose
      ScanArgs a{};
      a.R = BB * KB; a.S = TT; a.rpo = KB;
      a.x = z; a.x_outer = (long long)TT * KB * NB; a.x_inner = NB; a.x_step = (long long)KB * NB;
      a.wih = mods[j].d[0][0].wih; a.whh = mods[j].d[0][0].whh;
      a.bih = mods[j].d[0][0].bih; a.bhh = mods[j].d[0][0].bhh;
      a.out = l1; a.outD = NB; a.outOff = 0; a.rev = 0;
      lstm_scan<64><<<1, 128, 0, stream>>>(a);
      a.x = l1; a.x_outer = 0; a.x_inner = (long long)TT * NB; a.x_step = NB; a.rpo = BB * KB;
      a.wih = mods[j].d[1][0].wih; a.whh = mods[j].d[1][0].whh;
      a.bih = mods[j].d[1][0].bih; a.bhh = mods[j].d[1][0].bhh;
      a.out = l2;
      lstm_scan<64><<<1, 128, 0, stream>>>(a);
      fc_kernel<<<fcgrid, 128, 0, stream>>>(l2, mods[j].fw, mods[j].fb, hc, hn, MTOT, NB, 1);
    }
    bf16* tmp = hc; hc = hn; hn = tmp;
  }

  band_back_kernel<<<bgrid, 64, 0, stream>>>(hc, bbk, mr, mi);
  const int n = BB * FREQ * TT;
  final_mix<<<(n + 255) / 256, 256, 0, stream>>>(xr, xi, mr, mi, (float*)d_out, n);
}
